// MultiHeadHAR_61125974557567
// MI455X (gfx1250) — matrix operand order repaired and checked by reading, not compiled, not run on hardware
//
#include <hip/hip_runtime.h>

// ---------------------------------------------------------------------------
// Problem constants (from reference): B=128, T=256, D=6, H=512, NC=4, PS=12
// ---------------------------------------------------------------------------
#define Bsz   128
#define Tlen  256
#define Ddim  6
#define Hdim  512
#define NCc   4
#define PSn   12

#define NWG   64          // persistent workgroups; WG w owns h-cols [w*8, w*8+8)
#define NTH   256         // 8 wave32s per WG
#define JW    8           // h columns per WG
#define NCOLP 32          // packed gate columns per WG (4 gates * JW)
#define KC    64          // K staging chunk (bf16 elems) through LDS
#define HS    (Bsz*Hdim)  // elements in one h slot

typedef __bf16 bf16;
typedef __attribute__((ext_vector_type(16))) __bf16 v16bf;
typedef __attribute__((ext_vector_type(8)))  float  v8f;

union FragU { v16bf v; uint4 q[2]; };

__device__ inline v8f zero8() {
  v8f v;
#pragma unroll
  for (int i = 0; i < 8; ++i) v[i] = 0.f;
  return v;
}

__device__ inline float sigmoidf_(float v) { return 1.f / (1.f + __expf(-v)); }

// ---------------------------------------------------------------------------
// CDNA5 async global->LDS copy (ASYNCcnt-tracked, no VGPR round trip).
// VDST operand = 32-bit wave-relative LDS byte address (addrspace(3) value),
// VADDR = 64-bit global address, "off" = no SADDR (GV mode).
// ---------------------------------------------------------------------------
__device__ inline unsigned lds_addr_u32(const void* p) {
  return (unsigned)(uintptr_t)(__attribute__((address_space(3))) const char*)p;
}

__device__ inline void async_ld16(void* lds_dst, const void* gsrc) {
  asm volatile("global_load_async_to_lds_b128 %0, %1, off"
               :
               : "v"(lds_addr_u32(lds_dst)),
                 "v"((unsigned long long)(uintptr_t)gsrc)
               : "memory");
}

__device__ inline void async_wait0() {
  asm volatile("s_wait_asynccnt 0x0" ::: "memory");
}

// ---------------------------------------------------------------------------
// WMMA fragment loaders (LDS -> VGPR), CDNA5 wave32 16x16x32 bf16 layouts.
// A (16x32, MxK): lanes 0-15 row M=lane, K {0..7, 16..23}; lanes 16-31 same
// rows, K {8..15, 24..31}.  B (32x16, KxN): lanes 0-15 col N=lane K {0..7, 16..23},
// lanes 16-31 col N=lane-16 K {8..15, 24..31}.
// ---------------------------------------------------------------------------
__device__ inline v16bf load_afrag(const bf16* __restrict__ a_lds, int m0, int kc) {
  const int lane = threadIdx.x & 31;
  const int half = lane >> 4;
  const int r    = m0 + (lane & 15);
  const bf16* p  = a_lds + r * KC + kc + (half << 3);
  FragU f;
  f.q[0] = *(const uint4*)(p);        // K run 0 (8 bf16)
  f.q[1] = *(const uint4*)(p + 16);   // K run 1 (8 bf16)
  return f.v;
}

__device__ inline v16bf load_bfrag(const bf16* __restrict__ w_lds, int ktot, int n0, int k0) {
  const int lane = threadIdx.x & 31;
  const int col  = n0 + (lane & 15);
  const int ks   = k0 + ((lane >> 4) << 3);
  const bf16* p  = w_lds + (size_t)col * ktot + ks;
  FragU f;
  f.q[0] = *(const uint4*)(p);
  f.q[1] = *(const uint4*)(p + 16);
  return f.v;
}

// C/D f32 16x16: lanes 0-15 col N=lane rows m0+v; lanes 16-31 col N=lane-16 rows m0+8+v
__device__ inline void store_g(float* __restrict__ g_lds, v8f a, int m0, int n0) {
  const int lane = threadIdx.x & 31;
  const int col  = n0 + (lane & 15);
  const int mb   = m0 + ((lane >> 4) << 3);
#pragma unroll
  for (int v = 0; v < 8; ++v) g_lds[(mb + v) * NCOLP + col] = a[v];
}

// Stage one 128xKC bf16 activation chunk into an LDS buffer with async DMA.
// Thread copies 64B (4 x b128).  Chunk k-range maps to srcA (k<512) / srcB.
__device__ inline void stage_chunk_async(bf16* __restrict__ buf,
                                         const bf16* __restrict__ srcA,
                                         const bf16* __restrict__ srcB,
                                         int ch, int b, int off) {
  const int kabs = ch * KC + off;
  const bf16* src = (srcB != nullptr && kabs >= Hdim)
                        ? (srcB + (size_t)b * Hdim + (kabs - Hdim))
                        : (srcA + (size_t)b * Hdim + kabs);
  bf16* dst = buf + b * KC + off;
#pragma unroll
  for (int i = 0; i < 4; ++i) async_ld16(dst + i * 8, src + i * 8);
}

// ---------------------------------------------------------------------------
// One LSTM cell GEMM:  g(128 x 32cols) = cat(srcA, srcB)(128 x ktot) * W^T
// Weights resident in LDS (32 packed cols x ktot, bf16).  Activations staged
// through a double-buffered LDS chunk pair: the async DMA for chunk ch+1
// overlaps the WMMAs of chunk ch.  Each wave: 1 M-tile, 2 N-tiles.
// ---------------------------------------------------------------------------
__device__ inline void cell_gemm(const bf16* __restrict__ srcA,   // (128,512) bf16
                                 const bf16* __restrict__ srcB,   // (128,512) bf16 or null
                                 const bf16* __restrict__ w_lds,
                                 bf16* __restrict__ a_lds,        // 2 * 128*KC bf16
                                 int ktot, v8f* acc0, v8f* acc1) {
  const int tid = threadIdx.x;
  const int m0  = (tid >> 5) << 4;       // wave's M tile
  const int b   = tid >> 1;              // staging row
  const int off = (tid & 1) * 32;        // staging col offset within chunk
  const int nchunks = ktot / KC;
  *acc0 = zero8();
  *acc1 = zero8();

  __syncthreads();                       // previous consumers of a_lds done
  stage_chunk_async(a_lds, srcA, srcB, 0, b, off);
  async_wait0();
  __syncthreads();

  for (int ch = 0; ch < nchunks; ++ch) {
    const bf16* cur = a_lds + (size_t)(ch & 1) * (Bsz * KC);
    if (ch + 1 < nchunks)
      stage_chunk_async(a_lds + (size_t)((ch + 1) & 1) * (Bsz * KC),
                        srcA, srcB, ch + 1, b, off);
#pragma unroll
    for (int kc = 0; kc < KC; kc += 32) {
      const v16bf af = load_afrag(cur, m0, kc);
      const int k0 = ch * KC + kc;
      const v16bf b0 = load_bfrag(w_lds, ktot, 0,  k0);
      const v16bf b1 = load_bfrag(w_lds, ktot, 16, k0);
      *acc0 = __builtin_amdgcn_wmma_f32_16x16x32_bf16(false, af, false, b0,
                                                      (short)0, *acc0, false, false);
      *acc1 = __builtin_amdgcn_wmma_f32_16x16x32_bf16(false, af, false, b1,
                                                      (short)0, *acc1, false, false);
    }
    async_wait0();                       // next chunk's DMA landed
    __syncthreads();                     // + all waves done reading cur
  }
}

// ---------------------------------------------------------------------------
// Elementwise LSTM gates.  Thread owns (b = tid/2, 4 h-cols), keeps c in regs.
// ---------------------------------------------------------------------------
__device__ inline void cell_elem(const float* __restrict__ g_lds,
                                 const float bsum[16],
                                 const float* __restrict__ xW,   // (2048,6) fp32 or null
                                 const float* __restrict__ x,    // (B,T,6) fp32 or null
                                 int t,
                                 float c[4],
                                 bf16* __restrict__ h_out,       // (B,H) bf16
                                 bf16* __restrict__ out_t,       // (B,H) bf16 or null
                                 float* __restrict__ c_out) {    // (B,H) f32 or null
  const int tid = threadIdx.x;
  const int w   = blockIdx.x;
  const int b   = tid >> 1;
  const int j0  = (tid & 1) * 4;
  float xv[Ddim];
  if (x != nullptr) {
#pragma unroll
    for (int i = 0; i < Ddim; ++i) xv[i] = x[((size_t)b * Tlen + t) * Ddim + i];
  }
#pragma unroll
  for (int jj = 0; jj < 4; ++jj) {
    const int j = j0 + jj;
    float gv[4];
#pragma unroll
    for (int gate = 0; gate < 4; ++gate) {
      float g = g_lds[b * NCOLP + gate * JW + j] + bsum[gate * 4 + jj];
      if (x != nullptr) {
        const int gcol = gate * Hdim + w * JW + j;
#pragma unroll
        for (int i = 0; i < Ddim; ++i) g += xv[i] * xW[gcol * Ddim + i];
      }
      gv[gate] = g;
    }
    const float ig = sigmoidf_(gv[0]);
    const float fg = sigmoidf_(gv[1]);
    const float gg = tanhf(gv[2]);
    const float og = sigmoidf_(gv[3]);
    const float cn = fg * c[jj] + ig * gg;
    const float hn = og * tanhf(cn);
    c[jj] = cn;
    const int hcol = w * JW + j;
    h_out[(size_t)b * Hdim + hcol] = (bf16)hn;
    if (out_t != nullptr) out_t[(size_t)b * Hdim + hcol] = (bf16)hn;
    if (c_out != nullptr) c_out[(size_t)b * Hdim + hcol] = cn;
  }
}

// Device-wide barrier between recurrent steps (persistent kernel).
__device__ inline void grid_sync(unsigned* cnt, unsigned target) {
  __threadfence();
  __syncthreads();
  if (threadIdx.x == 0) {
    __hip_atomic_fetch_add(cnt, 1u, __ATOMIC_RELEASE, __HIP_MEMORY_SCOPE_AGENT);
    while (__hip_atomic_load(cnt, __ATOMIC_ACQUIRE, __HIP_MEMORY_SCOPE_AGENT) < target)
      __builtin_amdgcn_s_sleep(2);
  }
  __syncthreads();
}

// ---------------------------------------------------------------------------
// Kernels
// ---------------------------------------------------------------------------

// Zero grid-sync counters and both h double-buffers (h0 = 0 initial state).
__global__ void har_init_kernel(unsigned* __restrict__ cnt,
                                unsigned* __restrict__ h0u,   // 2*HS bf16 = 65536 u32
                                unsigned* __restrict__ h1u) {
  const int i = blockIdx.x * NTH + threadIdx.x;
  if (i < 8) cnt[i] = 0u;
  if (i < 65536) h0u[i] = 0u;
  else if (i < 131072) h1u[i - 65536] = 0u;
}

// Pack cat(WihA | WhhB) (each (2048, K*) fp32, gate-major rows) into bf16 rows
// grouped per owning WG: packed row r = w*32 + gate*8 + jl  <-  global row
// gate*512 + w*8 + jl.  KB==0 -> srcB unused (recurrent-only matrix).
__global__ void har_pack_kernel(bf16* __restrict__ dst,
                                const float* __restrict__ srcA,
                                const float* __restrict__ srcB,
                                int KA, int KB) {
  const int ktot = KA + KB;
  const long total = (long)4 * Hdim * ktot;
  const long idx = (long)blockIdx.x * NTH + threadIdx.x;
  if (idx >= total) return;
  const int r = (int)(idx / ktot);
  const int k = (int)(idx % ktot);
  const int w = r >> 5, p = r & 31, gate = p >> 3, jl = p & 7;
  const int gr = gate * Hdim + w * JW + jl;
  const float v = (k < KA) ? srcA[(size_t)gr * KA + k]
                           : srcB[(size_t)gr * KB + (k - KA)];
  dst[idx] = (bf16)v;
}

// Persistent encoder LSTM layer.  mode0 (layer0): inp_seq==null, K=512 over h,
// K=6 input term via VALU.  mode1 (layer1): A = cat(out0[t], h), K=1024.
__global__ void __launch_bounds__(NTH)
har_lstm_layer_kernel(const float* __restrict__ x,        // (B,T,6) or null
                      const float* __restrict__ Wih0,     // (2048,6) or null
                      const bf16*  __restrict__ inp_seq,  // (T,B,H) bf16 or null
                      const bf16*  __restrict__ Wp,       // packed (2048, ktot) bf16
                      const float* __restrict__ bih,
                      const float* __restrict__ bhh,
                      bf16*  __restrict__ hbuf,           // (2,B,H) bf16
                      float* __restrict__ cbuf,           // (B,H) f32, final c
                      bf16*  __restrict__ out_seq,        // (T,B,H) bf16 or null
                      unsigned* __restrict__ cnt,
                      int ktot) {
  extern __shared__ __align__(16) char smem[];
  bf16*  w_lds = (bf16*)smem;                                       // 32*ktot
  bf16*  a_lds = (bf16*)(smem + (size_t)NCOLP * ktot * 2);          // 2*128*KC
  float* g_lds = (float*)(smem + (size_t)NCOLP * ktot * 2 + 2 * Bsz * KC * 2);

  const int tid = threadIdx.x;
  const int w   = blockIdx.x;

  // Resident weights: this WG's 32 packed gate rows, full K, for all 256 steps.
  {
    const bf16* src = Wp + (size_t)w * NCOLP * ktot;
    const int total = NCOLP * ktot;
    for (int i = tid * 8; i < total; i += NTH * 8)
      async_ld16(w_lds + i, src + i);
    async_wait0();
    __syncthreads();
  }

  float bsum[16];
  {
    const int j0 = (tid & 1) * 4;
#pragma unroll
    for (int gate = 0; gate < 4; ++gate)
#pragma unroll
      for (int jj = 0; jj < 4; ++jj) {
        const int gcol = gate * Hdim + w * JW + j0 + jj;
        bsum[gate * 4 + jj] = bih[gcol] + bhh[gcol];
      }
  }

  float c[4] = {0.f, 0.f, 0.f, 0.f};
  const int m0 = (tid >> 5) << 4;

  for (int t = 0; t < Tlen; ++t) {
    const int rs = t & 1;
    const bf16* hread  = hbuf + (size_t)rs * HS;
    bf16*       hwrite = hbuf + (size_t)(1 - rs) * HS;
    const bf16* srcA = (inp_seq != nullptr) ? (inp_seq + (size_t)t * HS) : hread;
    const bf16* srcB = (inp_seq != nullptr) ? hread : nullptr;

    v8f acc0, acc1;
    cell_gemm(srcA, srcB, w_lds, a_lds, ktot, &acc0, &acc1);
    store_g(g_lds, acc0, m0, 0);
    store_g(g_lds, acc1, m0, 16);
    __syncthreads();

    cell_elem(g_lds, bsum, Wih0, x, t, c, hwrite,
              (out_seq != nullptr) ? (out_seq + (size_t)t * HS) : nullptr,
              (t == Tlen - 1) ? cbuf : nullptr);

    grid_sync(cnt, (unsigned)(NWG * (t + 1)));
  }
}

// Classifier + regression heads on last = out1[:, -1] (h1 final, slot 0).
__global__ void har_heads_kernel(const bf16*  __restrict__ last,   // (B,H) bf16
                                 const float* __restrict__ clsW,   // (4,512)
                                 const float* __restrict__ clsb,
                                 const float* __restrict__ regW,   // (1,512)
                                 const float* __restrict__ regb,
                                 float* __restrict__ out) {        // cls(512) | reg(128)
  const int o = blockIdx.x * NTH + threadIdx.x;
  if (o < Bsz * NCc) {
    const int b = o >> 2, cix = o & 3;
    float a = clsb[cix];
    for (int j = 0; j < Hdim; ++j)
      a += (float)last[(size_t)b * Hdim + j] * clsW[(size_t)cix * Hdim + j];
    out[o] = a;
  } else if (o < Bsz * NCc + Bsz) {
    const int b = o - Bsz * NCc;
    float a = regb[0];
    for (int j = 0; j < Hdim; ++j)
      a += (float)last[(size_t)b * Hdim + j] * regW[j];
    out[Bsz * NCc + b] = a;
  }
}

// Persistent 2-layer decoder, 12 steps.  Both cells' weight slices live in LDS.
__global__ void __launch_bounds__(NTH)
har_decoder_kernel(const bf16* __restrict__ D0P,   // packed cat(dec_Wih0|dec_Whh0)
                   const bf16* __restrict__ D1P,   // packed cat(dec_Wih1|dec_Whh1)
                   const float* __restrict__ bih0, const float* __restrict__ bhh0,
                   const float* __restrict__ bih1, const float* __restrict__ bhh1,
                   bf16* __restrict__ hbuf0,       // (2,B,H) continues encoder l0
                   bf16* __restrict__ hbuf1,       // (2,B,H) continues encoder l1
                   const float* __restrict__ c0i,  // (B,H) final c of enc l0
                   const float* __restrict__ c1i,  // (B,H) final c of enc l1
                   const float* __restrict__ fcW,  // (6,512)
                   const float* __restrict__ fcb,
                   float* __restrict__ forecast,   // (B,PS,6)
                   unsigned* __restrict__ cnt) {
  extern __shared__ __align__(16) char smem[];
  const int KT = 2 * Hdim;                         // 1024
  bf16*  w0    = (bf16*)smem;                      // 32*1024
  bf16*  w1    = w0 + (size_t)NCOLP * KT;
  bf16*  a_lds = w1 + (size_t)NCOLP * KT;          // 2*128*KC
  float* g_lds = (float*)(a_lds + 2 * Bsz * KC);

  const int tid = threadIdx.x;
  const int w   = blockIdx.x;

  {
    const bf16* s0 = D0P + (size_t)w * NCOLP * KT;
    const bf16* s1 = D1P + (size_t)w * NCOLP * KT;
    const int total = NCOLP * KT;
    for (int i = tid * 8; i < total; i += NTH * 8) {
      async_ld16(w0 + i, s0 + i);
      async_ld16(w1 + i, s1 + i);
    }
    async_wait0();
    __syncthreads();
  }

  float b0s[16], b1s[16];
  {
    const int j0 = (tid & 1) * 4;
#pragma unroll
    for (int gate = 0; gate < 4; ++gate)
#pragma unroll
      for (int jj = 0; jj < 4; ++jj) {
        const int gcol = gate * Hdim + w * JW + j0 + jj;
        b0s[gate * 4 + jj] = bih0[gcol] + bhh0[gcol];
        b1s[gate * 4 + jj] = bih1[gcol] + bhh1[gcol];
      }
  }

  float c0[4], c1[4];
  {
    const int b = tid >> 1, j0 = (tid & 1) * 4;
#pragma unroll
    for (int jj = 0; jj < 4; ++jj) {
      c0[jj] = c0i[(size_t)b * Hdim + w * JW + j0 + jj];
      c1[jj] = c1i[(size_t)b * Hdim + w * JW + j0 + jj];
    }
  }

  const int m0 = (tid >> 5) << 4;
  unsigned sidx = 0;

  for (int s = 0; s < PSn; ++s) {
    const int rs = s & 1;
    v8f acc0, acc1;

    // cell 0:  g0 = cat(inp = h1_prev, h0_prev) @ [Wih0|Whh0]^T
    cell_gemm(hbuf1 + (size_t)rs * HS, hbuf0 + (size_t)rs * HS,
              w0, a_lds, KT, &acc0, &acc1);
    store_g(g_lds, acc0, m0, 0);
    store_g(g_lds, acc1, m0, 16);
    __syncthreads();
    cell_elem(g_lds, b0s, nullptr, nullptr, 0, c0,
              hbuf0 + (size_t)(1 - rs) * HS, nullptr, nullptr);
    grid_sync(cnt, (unsigned)(NWG * (++sidx)));

    // cell 1:  g1 = cat(h0_new, h1_prev) @ [Wih1|Whh1]^T
    cell_gemm(hbuf0 + (size_t)(1 - rs) * HS, hbuf1 + (size_t)rs * HS,
              w1, a_lds, KT, &acc0, &acc1);
    store_g(g_lds, acc0, m0, 0);
    store_g(g_lds, acc1, m0, 16);
    __syncthreads();
    cell_elem(g_lds, b1s, nullptr, nullptr, 0, c1,
              hbuf1 + (size_t)(1 - rs) * HS, nullptr, nullptr);
    grid_sync(cnt, (unsigned)(NWG * (++sidx)));

    // forecast[:, s, :] = h1_new @ fc_W^T + fc_b  (last WG; double-buffering +
    // the barrier chain guarantees hbuf1[1-rs] survives until this finishes)
    if (w == NWG - 1) {
      const bf16* h1n = hbuf1 + (size_t)(1 - rs) * HS;
      for (int o = tid; o < Bsz * Ddim; o += NTH) {
        const int b = o / Ddim, d = o % Ddim;
        float a = fcb[d];
        for (int j = 0; j < Hdim; ++j)
          a += (float)h1n[(size_t)b * Hdim + j] * fcW[(size_t)d * Hdim + j];
        forecast[(size_t)b * (PSn * Ddim) + s * Ddim + d] = a;
      }
    }
  }
}

// ---------------------------------------------------------------------------
// Host launcher
// ---------------------------------------------------------------------------
extern "C" void kernel_launch(void* const* d_in, const int* in_sizes, int n_in,
                              void* d_out, int out_size, void* d_ws, size_t ws_size,
                              hipStream_t stream) {
  (void)in_sizes; (void)n_in; (void)out_size; (void)ws_size;
  const float* x        = (const float*)d_in[0];
  const float* eWih0    = (const float*)d_in[1];
  const float* eWhh0    = (const float*)d_in[2];
  const float* ebih0    = (const float*)d_in[3];
  const float* ebhh0    = (const float*)d_in[4];
  const float* eWih1    = (const float*)d_in[5];
  const float* eWhh1    = (const float*)d_in[6];
  const float* ebih1    = (const float*)d_in[7];
  const float* ebhh1    = (const float*)d_in[8];
  const float* dWih0    = (const float*)d_in[9];
  const float* dWhh0    = (const float*)d_in[10];
  const float* dbih0    = (const float*)d_in[11];
  const float* dbhh0    = (const float*)d_in[12];
  const float* dWih1    = (const float*)d_in[13];
  const float* dWhh1    = (const float*)d_in[14];
  const float* dbih1    = (const float*)d_in[15];
  const float* dbhh1    = (const float*)d_in[16];
  const float* fcW      = (const float*)d_in[17];
  const float* fcb      = (const float*)d_in[18];
  const float* clsW     = (const float*)d_in[19];
  const float* clsb     = (const float*)d_in[20];
  const float* regW     = (const float*)d_in[21];
  const float* regb     = (const float*)d_in[22];
  float* outp = (float*)d_out;

  // Workspace carve-up
  char* ws = (char*)d_ws;
  unsigned* cnt = (unsigned*)ws;
  size_t off = 256;
  bf16*  hbuf0 = (bf16*)(ws + off);  off += (size_t)2 * HS * sizeof(bf16);
  bf16*  hbuf1 = (bf16*)(ws + off);  off += (size_t)2 * HS * sizeof(bf16);
  float* cbuf0 = (float*)(ws + off); off += (size_t)HS * sizeof(float);
  float* cbuf1 = (float*)(ws + off); off += (size_t)HS * sizeof(float);
  bf16*  W0P   = (bf16*)(ws + off);  off += (size_t)4 * Hdim * Hdim * sizeof(bf16);
  bf16*  W1P   = (bf16*)(ws + off);  off += (size_t)4 * Hdim * 2 * Hdim * sizeof(bf16);
  bf16*  D0P   = (bf16*)(ws + off);  off += (size_t)4 * Hdim * 2 * Hdim * sizeof(bf16);
  bf16*  D1P   = (bf16*)(ws + off);  off += (size_t)4 * Hdim * 2 * Hdim * sizeof(bf16);
  bf16*  out0  = (bf16*)(ws + off);  off += (size_t)Tlen * HS * sizeof(bf16);

  // 1) init counters + zero h state
  har_init_kernel<<<512, NTH, 0, stream>>>(cnt, (unsigned*)hbuf0, (unsigned*)hbuf1);

  // 2) pack weights to bf16 WMMA-friendly per-WG slices
  {
    const long tot0 = (long)4 * Hdim * Hdim;            // 2048*512
    const long tot1 = (long)4 * Hdim * 2 * Hdim;        // 2048*1024
    har_pack_kernel<<<(unsigned)((tot0 + NTH - 1) / NTH), NTH, 0, stream>>>(W0P, eWhh0, nullptr, Hdim, 0);
    har_pack_kernel<<<(unsigned)((tot1 + NTH - 1) / NTH), NTH, 0, stream>>>(W1P, eWih1, eWhh1, Hdim, Hdim);
    har_pack_kernel<<<(unsigned)((tot1 + NTH - 1) / NTH), NTH, 0, stream>>>(D0P, dWih0, dWhh0, Hdim, Hdim);
    har_pack_kernel<<<(unsigned)((tot1 + NTH - 1) / NTH), NTH, 0, stream>>>(D1P, dWih1, dWhh1, Hdim, Hdim);
  }

  // 3) encoder layer 0 (persistent, K=512 recurrent + K=6 VALU input term)
  {
    const size_t sm = (size_t)NCOLP * Hdim * 2 + 2 * Bsz * KC * 2 + Bsz * NCOLP * 4; // 80 KB
    har_lstm_layer_kernel<<<NWG, NTH, sm, stream>>>(
        x, eWih0, nullptr, W0P, ebih0, ebhh0, hbuf0, cbuf0, out0, cnt + 0, Hdim);
  }

  // 4) encoder layer 1 (persistent, K=1024: cat(out0[t], h))
  {
    const size_t sm = (size_t)NCOLP * 2 * Hdim * 2 + 2 * Bsz * KC * 2 + Bsz * NCOLP * 4; // 112 KB
    har_lstm_layer_kernel<<<NWG, NTH, sm, stream>>>(
        nullptr, nullptr, out0, W1P, ebih1, ebhh1, hbuf1, cbuf1, nullptr, cnt + 1, 2 * Hdim);
  }

  // 5) heads on last = h1 final (slot 0); must run before the decoder mutates hbuf1
  har_heads_kernel<<<3, NTH, 0, stream>>>(
      hbuf1, clsW, clsb, regW, regb, outp + (size_t)Bsz * PSn * Ddim);

  // 6) decoder (persistent, 12 steps x 2 cells, K=1024 each)
  {
    const size_t sm = (size_t)2 * NCOLP * 2 * Hdim * 2 + 2 * Bsz * KC * 2 + Bsz * NCOLP * 4; // 176 KB
    har_decoder_kernel<<<NWG, NTH, sm, stream>>>(
        D0P, D1P, dbih0, dbhh0, dbih1, dbhh1,
        hbuf0, hbuf1, cbuf0, cbuf1, fcW, fcb, outp, cnt + 2);
  }
}
